// VisualPacker_3d_20083267076714
// MI455X (gfx1250) — matrix operand order repaired and checked by reading, not compiled, not run on hardware
//
#include <hip/hip_runtime.h>
#include <hip/hip_bf16.h>

typedef __attribute__((ext_vector_type(16))) __bf16 v16bf;
typedef __attribute__((ext_vector_type(8)))  float  v8f;
typedef __attribute__((ext_vector_type(4)))  unsigned int u32x4;
typedef __attribute__((ext_vector_type(8)))  unsigned int u32x8;

// ---------------------------------------------------------------------------
// fp32 [K,N] -> bf16 transposed [N,K]  (tiled 32x32 through LDS, coalesced)
// ---------------------------------------------------------------------------
__global__ __launch_bounds__(256) void f32_to_bf16_transpose_kernel(const float* __restrict__ src,
                                                                    __hip_bfloat16* __restrict__ dst,
                                                                    int K, int N) {
    __shared__ float tile[32][33];
    const int tx = threadIdx.x & 31;
    const int ty = threadIdx.x >> 5;          // 0..7
    const int nb = blockIdx.x * 32;
    const int kb = blockIdx.y * 32;
#pragma unroll
    for (int j = 0; j < 4; ++j)
        tile[ty + 8 * j][tx] = src[(size_t)(kb + ty + 8 * j) * N + nb + tx];
    __syncthreads();
#pragma unroll
    for (int j = 0; j < 4; ++j)
        dst[(size_t)(nb + ty + 8 * j) * K + kb + tx] = __float2bfloat16(tile[tx][ty + 8 * j]);
}

// ---------------------------------------------------------------------------
// Local cross-resolution attention absorb:
//   visual [32,2048,768] f32 -> absorbed [8192,768] bf16
// One 256-thread block per LR token; thread t owns dims {t, t+256, t+512}.
// ---------------------------------------------------------------------------
__global__ __launch_bounds__(256) void attn_absorb_kernel(const float* __restrict__ vis,
                                                          __hip_bfloat16* __restrict__ absorbed) {
    const int t   = blockIdx.x;        // 0..8191  (b*256 + x*64 + y*8 + z)
    const int tid = threadIdx.x;       // 0..255
    const int b   = t >> 8;
    const int rem = t & 255;
    const int x   = rem >> 6;
    const int y   = (rem >> 3) & 7;
    const int z   = rem & 7;
    const float* base = vis + (size_t)b * 2048 * 768;

    int hr[8];
#pragma unroll
    for (int k = 0; k < 8; ++k) {
        const int i = (k >> 2) & 1, j = (k >> 1) & 1, l = k & 1;
        hr[k] = ((2 * x + i) * 256 + (2 * y + j) * 16 + (2 * z + l)) * 768;
    }

    float vals[3][8];
    float p[8];
#pragma unroll
    for (int k = 0; k < 8; ++k) p[k] = 0.f;

#pragma unroll
    for (int r = 0; r < 3; ++r) {
        const int d = tid + 256 * r;
        float s = 0.f;
#pragma unroll
        for (int k = 0; k < 8; ++k) { vals[r][k] = base[hr[k] + d]; s += vals[r][k]; }
        const float lr = s * 0.125f;               // avg-pool = mean over 8 HR points
#pragma unroll
        for (int k = 0; k < 8; ++k) p[k] += lr * vals[r][k];
    }

    // wave32 reduction of the 8 partial dot products, then cross-wave via LDS
#pragma unroll
    for (int k = 0; k < 8; ++k)
#pragma unroll
        for (int off = 16; off; off >>= 1)
            p[k] += __shfl_xor(p[k], off, 32);

    __shared__ float sred[8 * 8];
    const int wid = tid >> 5, lane = tid & 31;
    if (lane == 0) {
#pragma unroll
        for (int k = 0; k < 8; ++k) sred[wid * 8 + k] = p[k];
    }
    __syncthreads();

    float score[8];
#pragma unroll
    for (int k = 0; k < 8; ++k) {
        float s = 0.f;
#pragma unroll
        for (int w = 0; w < 8; ++w) s += sred[w * 8 + k];
        score[k] = s * 0.03608439182435161f;       // 1/sqrt(768)
    }
    float mx = score[0];
#pragma unroll
    for (int k = 1; k < 8; ++k) mx = fmaxf(mx, score[k]);
    float e[8], den = 0.f;
#pragma unroll
    for (int k = 0; k < 8; ++k) { e[k] = __expf(score[k] - mx); den += e[k]; }
    const float inv = 1.f / den;

#pragma unroll
    for (int r = 0; r < 3; ++r) {
        const int d = tid + 256 * r;
        float acc = 0.f;
#pragma unroll
        for (int k = 0; k < 8; ++k) acc += (e[k] * inv) * vals[r][k];
        absorbed[(size_t)t * 768 + d] = __float2bfloat16(acc);
    }
}

// ---------------------------------------------------------------------------
// TDM: async 2D tile load global -> LDS (128 rows x 32 bf16, padded rows).
// D# group0: count=1 | lds_addr | global_addr | type=2
// D# group1: data_size=2B, pad_enable (4 DWORDs every 16 DWORDs -> 80B rows),
//            tensor_dim0=K, tensor_dim1=rows, tile=32x128, dim0_stride=K.
// Issued by one wave; completion tracked with TENSORcnt.
// ---------------------------------------------------------------------------
__device__ __forceinline__ void tdm_load_tile_2d(unsigned lds_off,
                                                 const void* gaddr,
                                                 unsigned k_elems,
                                                 unsigned rows_total) {
    const unsigned long long ga = (unsigned long long)gaddr;
    u32x4 g0;
    g0[0] = 1u;                                     // count=1, user mode
    g0[1] = lds_off;                                // lds_addr (bytes)
    g0[2] = (unsigned)ga;                           // global_addr[31:0]
    g0[3] = (unsigned)(ga >> 32) | (2u << 30);      // global_addr[56:32] | type=2

    u32x8 g1;
    g1[0] = (1u << 16)                              // data_size = 1 -> 2 bytes
          | (1u << 20)                              // pad_enable
          | (3u << 22)                              // pad_interval: 16 DWORDs (64B)
          | (3u << 25);                             // pad_amount: 4 DWORDs (16B)
    g1[1] = (k_elems & 0xffffu) << 16;              // tensor_dim0[15:0]
    g1[2] = ((k_elems >> 16) & 0xffffu)             // tensor_dim0[31:16]
          | ((rows_total & 0xffffu) << 16);         // tensor_dim1[15:0]
    g1[3] = ((rows_total >> 16) & 0xffffu)          // tensor_dim1[31:16]
          | (32u << 16);                            // tile_dim0 = 32
    g1[4] = 128u;                                   // tile_dim1 = 128, tile_dim2 = 0
    g1[5] = k_elems;                                // tensor_dim0_stride[31:0]
    g1[6] = 0u;                                     // stride[47:32], dim1_stride lo
    g1[7] = 0u;

    asm volatile("tensor_load_to_lds %0, %1" :: "s"(g0), "s"(g1) : "memory");
}

// ---------------------------------------------------------------------------
// bf16 WMMA GEMM:  C[M,N] = act( A[M,K] * Bt[N,K]^T + bias[N] )
// Block tile 128(M) x 128(N), 8 waves; wave tile 32x64 = 8 WMMA / K-step.
// LDS double-buffered; tiles staged by the Tensor Data Mover (wave 0 issues,
// waits TENSORcnt, then the workgroup barrier publishes the buffer).
// ---------------------------------------------------------------------------
template <bool GELU, typename OutT>
__global__ __launch_bounds__(256) void gemm_wmma_kernel(const __hip_bfloat16* __restrict__ A,
                                                        const __hip_bfloat16* __restrict__ Bt,
                                                        const float* __restrict__ bias,
                                                        OutT* __restrict__ C,
                                                        int M, int N, int K) {
    constexpr int LDT = 40;                       // 32 elems + 16B TDM pad per row
    __shared__ unsigned short sA[2][128 * LDT];
    __shared__ unsigned short sB[2][128 * LDT];

    const int tid  = threadIdx.x;
    const int lane = tid & 31;
    const int wid  = tid >> 5;
    const int lo   = lane & 15;                   // M row (A) / N col (B,C)
    const int hi   = lane >> 4;                   // lane-half selector
    const int waveM = (wid & 3) * 32;
    const int waveN = (wid >> 2) * 64;
    const int Mbase = blockIdx.y * 128;
    const int Nbase = blockIdx.x * 128;

    const unsigned short* gA = (const unsigned short*)A;
    const unsigned short* gB = (const unsigned short*)Bt;

    const unsigned ldsA0 = (unsigned)(unsigned long long)(void*)&sA[0][0];
    const unsigned ldsA1 = (unsigned)(unsigned long long)(void*)&sA[1][0];
    const unsigned ldsB0 = (unsigned)(unsigned long long)(void*)&sB[0][0];
    const unsigned ldsB1 = (unsigned)(unsigned long long)(void*)&sB[1][0];

    v8f acc[2][4];
#pragma unroll
    for (int m = 0; m < 2; ++m)
#pragma unroll
        for (int t = 0; t < 4; ++t)
            acc[m][t] = (v8f){0.f, 0.f, 0.f, 0.f, 0.f, 0.f, 0.f, 0.f};

    union Frag { uint4 u[2]; v16bf v; };

    // prologue: DMA first K-tile into buffer 0
    if (wid == 0) {
        tdm_load_tile_2d(ldsA0, gA + (size_t)Mbase * K, (unsigned)K, (unsigned)M);
        tdm_load_tile_2d(ldsB0, gB + (size_t)Nbase * K, (unsigned)K, (unsigned)N);
        __builtin_amdgcn_s_wait_tensorcnt(0);
    }
    __syncthreads();                              // buffer 0 published

    int buf = 0;
    for (int kb = 0; kb < K; kb += 32) {
        const bool more = (kb + 32) < K;
        if (more && wid == 0) {                   // DMA next tile into other buffer
            const unsigned la = buf ? ldsA0 : ldsA1;
            const unsigned lb = buf ? ldsB0 : ldsB1;
            tdm_load_tile_2d(la, gA + (size_t)Mbase * K + kb + 32, (unsigned)K, (unsigned)M);
            tdm_load_tile_2d(lb, gB + (size_t)Nbase * K + kb + 32, (unsigned)K, (unsigned)N);
        }

        const unsigned short* cA = &sA[buf][0];
        const unsigned short* cB = &sB[buf][0];

        // A fragments: lane lo = M row; elems 0..7 -> K=8*hi.., 8..15 -> K=16+8*hi..
        Frag fa[2];
#pragma unroll
        for (int m = 0; m < 2; ++m) {
            const int arow = waveM + m * 16 + lo;
            fa[m].u[0] = *(const uint4*)&cA[arow * LDT + hi * 8];
            fa[m].u[1] = *(const uint4*)&cA[arow * LDT + 16 + hi * 8];
        }
#pragma unroll
        for (int t = 0; t < 4; ++t) {
            // B fragment: lane lo = N col; elems 0..7 -> K=8*hi.., 8..15 -> K=16+8*hi..
            Frag fb;
            const int brow = waveN + t * 16 + lo;
            fb.u[0] = *(const uint4*)&cB[brow * LDT + hi * 8];
            fb.u[1] = *(const uint4*)&cB[brow * LDT + 16 + hi * 8];
#pragma unroll
            for (int m = 0; m < 2; ++m)
                acc[m][t] = __builtin_amdgcn_wmma_f32_16x16x32_bf16(
                    false, fa[m].v, false, fb.v, (short)0, acc[m][t], false, false);
        }

        if (more && wid == 0)
            __builtin_amdgcn_s_wait_tensorcnt(0); // next buffer complete
        __syncthreads();                          // publish next buffer / retire reads
        buf ^= 1;
    }

    // --- epilogue: bias (+ exact-erf GELU), store ---
#pragma unroll
    for (int t = 0; t < 4; ++t) {
        const int col = Nbase + waveN + t * 16 + lo;
        const float bc = bias[col];
#pragma unroll
        for (int m = 0; m < 2; ++m) {
#pragma unroll
            for (int i = 0; i < 8; ++i) {
                const int row = Mbase + waveM + m * 16 + hi * 8 + i;
                float v = acc[m][t][i] + bc;
                if constexpr (GELU)
                    v = 0.5f * v * (1.0f + erff(v * 0.70710678118654752f));
                if constexpr (sizeof(OutT) == 2)
                    ((__hip_bfloat16*)C)[(size_t)row * N + col] = __float2bfloat16(v);
                else
                    ((float*)C)[(size_t)row * N + col] = v;
            }
        }
    }
}

// ---------------------------------------------------------------------------
// Launch
// ---------------------------------------------------------------------------
extern "C" void kernel_launch(void* const* d_in, const int* in_sizes, int n_in,
                              void* d_out, int out_size, void* d_ws, size_t ws_size,
                              hipStream_t stream) {
    (void)in_sizes; (void)n_in; (void)out_size; (void)ws_size;
    const float* vis = (const float*)d_in[0];   // [32,2048,768]
    const float* W1  = (const float*)d_in[1];   // [768,4096]
    const float* b1  = (const float*)d_in[2];   // [4096]
    const float* W2  = (const float*)d_in[3];   // [4096,4096]
    const float* b2  = (const float*)d_in[4];   // [4096]
    float* out = (float*)d_out;                 // [32,256,4096]

    const int M = 8192, D = 768, O = 4096;

    char* ws = (char*)d_ws;
    size_t off = 0;
    auto take = [&](size_t bytes) {
        void* p = ws + off;
        off = (off + bytes + 255) & ~(size_t)255;
        return p;
    };
    __hip_bfloat16* absorbed = (__hip_bfloat16*)take((size_t)M * D * 2);   // 12.6 MB
    __hip_bfloat16* W1t      = (__hip_bfloat16*)take((size_t)D * O * 2);   //  6.3 MB  [O,D]
    __hip_bfloat16* W2t      = (__hip_bfloat16*)take((size_t)O * O * 2);   // 33.6 MB  [O,O]
    __hip_bfloat16* h        = (__hip_bfloat16*)take((size_t)M * O * 2);   // 67.1 MB

    // transpose weights to [N,K] bf16 so GEMM B-operand is K-contiguous
    f32_to_bf16_transpose_kernel<<<dim3(O / 32, D / 32), 256, 0, stream>>>(W1, W1t, D, O);
    f32_to_bf16_transpose_kernel<<<dim3(O / 32, O / 32), 256, 0, stream>>>(W2, W2t, O, O);

    attn_absorb_kernel<<<M, 256, 0, stream>>>(vis, absorbed);

    dim3 grid(O / 128, M / 128);   // (32, 64)
    gemm_wmma_kernel<true,  __hip_bfloat16><<<grid, 256, 0, stream>>>(absorbed, W1t, b1, h,   M, O, D);
    gemm_wmma_kernel<false, float         ><<<grid, 256, 0, stream>>>(h,        W2t, b2, out, M, O, O);
}
